// MAGNN_metapath_specific_35631048687986
// MI455X (gfx1250) — matrix operand order repaired and checked by reading, not compiled, not run on hardware
//
#include <hip/hip_runtime.h>
#include <hip/hip_bf16.h>

// ---------------------------------------------------------------------------
// MAGNN metapath-specific forward for MI455X (gfx1250, wave32, WMMA).
// Pipeline: init -> weight repack (fp32 -> bf16 B-fragment layout) ->
// fused GRU+attention (WMMA 16x16x32 bf16) -> segment softmax -> aggregate.
// ---------------------------------------------------------------------------

typedef __attribute__((ext_vector_type(16))) __bf16 v16bf;
typedef __attribute__((ext_vector_type(8)))  float  v8f;

#define HSTRIDE 528   // 512 + 16 pad (bf16 elems) -> conflict-free ds_load_b128
#define XSTRIDE 72    // 64 + 8 pad

__device__ __forceinline__ unsigned short f2bf(float f) {
    unsigned u = __float_as_uint(f);
    unsigned r = (u + 0x7FFFu + ((u >> 16) & 1u)) >> 16;   // round-to-nearest-even
    return (unsigned short)r;
}
__device__ __forceinline__ float bf2f(unsigned short s) {
    return __uint_as_float(((unsigned)s) << 16);
}
__device__ __forceinline__ float sigm(float x) { return 1.0f / (1.0f + __expf(-x)); }

// order-preserving float -> uint encoding for atomicMax
__device__ __forceinline__ unsigned fenc(float f) {
    unsigned u = __float_as_uint(f);
    return (u & 0x80000000u) ? ~u : (u | 0x80000000u);
}
__device__ __forceinline__ float fdec(unsigned u) {
    unsigned b = (u & 0x80000000u) ? (u & 0x7FFFFFFFu) : ~u;
    return __uint_as_float(b);
}

union Frag {
    uint4  q[2];
    v16bf  v;
};

// A fragment (16x32 bf16, M = lane%16, lanes>=16 start at K+8; VGPR4-7 = K+16)
__device__ __forceinline__ v16bf load_a(const unsigned short* lds, int rowStride,
                                        int rowBase, int kbase, int lane) {
    int m  = rowBase + (lane & 15);
    int k0 = kbase + ((lane & 16) ? 8 : 0);
    const unsigned short* p = lds + m * rowStride + k0;
    Frag f;
    f.q[0] = *(const uint4*)(p);
    f.q[1] = *(const uint4*)(p + 16);
    return f.v;
}

// B fragment: pre-packed so each lane reads 16 contiguous bf16 (32B)
__device__ __forceinline__ v16bf load_b(const unsigned short* pack, int ntile,
                                        int ktile, int nkt, int lane) {
    const unsigned short* p = pack + ((((size_t)ntile * nkt) + ktile) * 32 + lane) * 16;
    Frag f;
    f.q[0] = *(const uint4*)(p);
    f.q[1] = *(const uint4*)(p + 8);
    return f.v;
}

__device__ __forceinline__ v8f wmma_bf16(v16bf a, v16bf b, v8f c) {
    return __builtin_amdgcn_wmma_f32_16x16x32_bf16(false, a, false, b, (short)0, c,
                                                   false, false);
}

// ---------------------------------------------------------------------------
__global__ void init_kernel(float* __restrict__ out, unsigned* __restrict__ amax,
                            float* __restrict__ asum, int outN, int ndh) {
    int i = blockIdx.x * blockDim.x + threadIdx.x;
    if (i < outN) out[i] = 0.0f;
    if (i < ndh) { amax[i] = 0u; asum[i] = 0.0f; }
}

// repack W (row-major [N][K] fp32) into bf16 B-fragment order:
// idx = (((nt*nkt + kt)*32 + lane)*16 + v2), element (k,n):
//   n = nt*16 + lane%16 ; k = kt*32 + (lane&16)/2 + (v2&7) + (v2&8)*2
__global__ void repack_kernel(const float* __restrict__ W, unsigned short* __restrict__ P,
                              int K, int nkt, int total) {
    int i = blockIdx.x * blockDim.x + threadIdx.x;
    if (i >= total) return;
    int v2   = i & 15;
    int lane = (i >> 4) & 31;
    int kt   = (i >> 9) % nkt;
    int nt   = (i >> 9) / nkt;
    int n = nt * 16 + (lane & 15);
    int k = kt * 32 + ((lane & 16) >> 1) + (v2 & 7) + ((v2 & 8) << 1);
    P[i] = f2bf(W[(size_t)n * K + k]);
}

// ---------------------------------------------------------------------------
// Fused GRU (3 steps) + attention logits. Block = 32 edges, 256 threads (8 waves).
// Wave w owns hidden columns [64w, 64w+64). Hidden state double-buffered in LDS.
// ---------------------------------------------------------------------------
__global__ __launch_bounds__(256) void gru_attn_kernel(
    const float* __restrict__ features,
    const unsigned short* __restrict__ WihP,   // 96 ntiles x 2 kt
    const unsigned short* __restrict__ WhhP,   // 96 ntiles x 16 kt
    const float* __restrict__ b_ih,
    const float* __restrict__ b_hh,
    const float* __restrict__ attn,            // [8][64]
    const int* __restrict__ emi,               // [E][L]
    const int* __restrict__ dstv,              // [E]
    unsigned short* __restrict__ Hq,           // [E][512] bf16
    float* __restrict__ a_buf,                 // [E][8]
    unsigned* __restrict__ amaxEnc,            // [ND][8]
    int E, int L)
{
    __shared__ __align__(16) unsigned short xB[3 * 32 * XSTRIDE];
    __shared__ __align__(16) unsigned short hB[2 * 32 * HSTRIDE];

    const int tid  = threadIdx.x;
    const int lane = tid & 31;
    const int w    = tid >> 5;
    const int e0   = blockIdx.x * 32;

    // ---- gather metapath features (3 steps x 32 edges x 64 dims) -> bf16 LDS
    for (int i = tid; i < 3 * 32 * 64; i += 256) {
        int t  = i >> 11;           // / (32*64)
        int r  = i & 2047;
        int el = r >> 6;
        int d  = r & 63;
        int e  = e0 + el; if (e >= E) e = E - 1;
        int node = emi[(size_t)e * L + t];
        xB[(t * 32 + el) * XSTRIDE + d] = f2bf(features[(size_t)node * 64 + d]);
    }
    __syncthreads();

    const int ncol = lane & 15;
    const int mh   = (lane >> 4) * 8;

    for (int t = 0; t < 3; ++t) {
        const unsigned short* hRead  = hB + ((t + 1) & 1) * 32 * HSTRIDE;
        unsigned short*       hWrite = hB + (t & 1) * 32 * HSTRIDE;
        const unsigned short* xbase  = xB + t * 32 * XSTRIDE;

        for (int c = 0; c < 4; ++c) {
            v8f r0 = {}, r1 = {}, z0 = {}, z1 = {};
            v8f ni0 = {}, ni1 = {}, nh0 = {}, nh1 = {};
            const int ntr = w * 4 + c;     // r-gate N-tile
            const int ntz = 32 + ntr;      // z-gate
            const int ntn = 64 + ntr;      // n-gate

            // x @ W_ih^T  (K = 64 -> 2 K-tiles)
            #pragma unroll
            for (int kt = 0; kt < 2; ++kt) {
                v16bf a0 = load_a(xbase, XSTRIDE, 0,  kt * 32, lane);
                v16bf a1 = load_a(xbase, XSTRIDE, 16, kt * 32, lane);
                v16bf br = load_b(WihP, ntr, kt, 2, lane);
                v16bf bz = load_b(WihP, ntz, kt, 2, lane);
                v16bf bn = load_b(WihP, ntn, kt, 2, lane);
                r0  = wmma_bf16(a0, br, r0);   r1  = wmma_bf16(a1, br, r1);
                z0  = wmma_bf16(a0, bz, z0);   z1  = wmma_bf16(a1, bz, z1);
                ni0 = wmma_bf16(a0, bn, ni0);  ni1 = wmma_bf16(a1, bn, ni1);
            }
            // h @ W_hh^T  (K = 512 -> 16 K-tiles); h == 0 at t == 0
            if (t > 0) {
                #pragma unroll 4
                for (int kt = 0; kt < 16; ++kt) {
                    v16bf a0 = load_a(hRead, HSTRIDE, 0,  kt * 32, lane);
                    v16bf a1 = load_a(hRead, HSTRIDE, 16, kt * 32, lane);
                    v16bf br = load_b(WhhP, ntr, kt, 16, lane);
                    v16bf bz = load_b(WhhP, ntz, kt, 16, lane);
                    v16bf bn = load_b(WhhP, ntn, kt, 16, lane);
                    r0  = wmma_bf16(a0, br, r0);   r1  = wmma_bf16(a1, br, r1);
                    z0  = wmma_bf16(a0, bz, z0);   z1  = wmma_bf16(a1, bz, z1);
                    nh0 = wmma_bf16(a0, bn, nh0);  nh1 = wmma_bf16(a1, bn, nh1);
                }
            }

            // gate math + h update (C-frag layout: VGPR v -> M = v + 8*(lane>=16))
            const int j = w * 64 + c * 16 + ncol;
            const float bir  = b_ih[j] + b_hh[j];
            const float biz  = b_ih[512 + j] + b_hh[512 + j];
            const float bin_ = b_ih[1024 + j];
            const float bhn  = b_hh[1024 + j];
            #pragma unroll
            for (int v = 0; v < 8; ++v) {
                {   // M-tile 0
                    int m = mh + v;
                    float rr = sigm(r0[v] + bir);
                    float zz = sigm(z0[v] + biz);
                    float nn = tanhf(ni0[v] + bin_ + rr * (nh0[v] + bhn));
                    float ho = (t == 0) ? 0.0f : bf2f(hRead[m * HSTRIDE + j]);
                    hWrite[m * HSTRIDE + j] = f2bf((1.0f - zz) * nn + zz * ho);
                }
                {   // M-tile 1
                    int m = 16 + mh + v;
                    float rr = sigm(r1[v] + bir);
                    float zz = sigm(z1[v] + biz);
                    float nn = tanhf(ni1[v] + bin_ + rr * (nh1[v] + bhn));
                    float ho = (t == 0) ? 0.0f : bf2f(hRead[m * HSTRIDE + j]);
                    hWrite[m * HSTRIDE + j] = f2bf((1.0f - zz) * nn + zz * ho);
                }
            }
        }
        __syncthreads();
    }

    // final hidden is in buffer 0 (t=2 wrote t&1 == 0)
    const unsigned short* hFin = hB;

    // attention logits: 256 threads -> (edge, head)
    {
        int el = tid >> 3, head = tid & 7;
        float s = 0.0f;
        #pragma unroll 8
        for (int d = 0; d < 64; ++d)
            s += bf2f(hFin[el * HSTRIDE + head * 64 + d]) * attn[head * 64 + d];
        float av = s > 0.0f ? s : 0.01f * s;   // leaky relu
        int e = e0 + el;
        if (e < E) {
            a_buf[(size_t)e * 8 + head] = av;
            atomicMax(&amaxEnc[(size_t)dstv[e] * 8 + head], fenc(av));
        }
    }
    // spill final hidden (bf16) for the aggregation pass
    for (int i = tid; i < 32 * 512; i += 256) {
        int el = i >> 9, jj = i & 511;
        int e = e0 + el;
        if (e < E) Hq[(size_t)e * 512 + jj] = hFin[el * HSTRIDE + jj];
    }
}

// ---------------------------------------------------------------------------
__global__ void exp_sum_kernel(const float* __restrict__ a_buf,
                               const unsigned* __restrict__ amaxEnc,
                               const int* __restrict__ dstv,
                               float* __restrict__ aexp, float* __restrict__ asum,
                               int EH) {
    int i = blockIdx.x * blockDim.x + threadIdx.x;
    if (i >= EH) return;
    int e = i >> 3, head = i & 7;
    int d0 = dstv[e];
    float m  = fdec(amaxEnc[(size_t)d0 * 8 + head]);
    float ex = __expf(a_buf[i] - m);
    aexp[i] = ex;
    atomicAdd(&asum[(size_t)d0 * 8 + head], ex);
}

__global__ void aggregate_kernel(const unsigned short* __restrict__ Hq,
                                 const float* __restrict__ aexp,
                                 const float* __restrict__ asum,
                                 const int* __restrict__ dstv,
                                 float* __restrict__ out, int E) {
    size_t i = (size_t)blockIdx.x * blockDim.x + threadIdx.x;
    if (i >= (size_t)E * 512) return;
    int e = (int)(i >> 9);
    int j = (int)(i & 511);
    int head = j >> 6;
    int d0 = dstv[e];
    float alpha = aexp[(size_t)e * 8 + head] / asum[(size_t)d0 * 8 + head];
    atomicAdd(&out[(size_t)d0 * 512 + j], alpha * bf2f(Hq[i]));
}

// ---------------------------------------------------------------------------
extern "C" void kernel_launch(void* const* d_in, const int* in_sizes, int n_in,
                              void* d_out, int out_size, void* d_ws, size_t ws_size,
                              hipStream_t stream) {
    const float* features = (const float*)d_in[0];
    const float* W_ih     = (const float*)d_in[1];
    const float* W_hh     = (const float*)d_in[2];
    const float* b_ih     = (const float*)d_in[3];
    const float* b_hh     = (const float*)d_in[4];
    const float* attn     = (const float*)d_in[5];
    const int*   emi      = (const int*)d_in[6];
    const int*   dstv     = (const int*)d_in[7];

    const int E  = in_sizes[7];
    const int L  = in_sizes[6] / E;          // == 3
    const int ND = out_size / 512;
    float* out = (float*)d_out;

    // workspace carving (256B aligned)
    size_t off = 0;
    char* base = (char*)d_ws;
    auto carve = [&](size_t bytes) -> void* {
        void* p = base + off;
        off += (bytes + 255) & ~(size_t)255;
        return p;
    };
    const size_t wihN = 96u * 2u * 32u * 16u;    // 98304  (1536x64)
    const size_t whhN = 96u * 16u * 32u * 16u;   // 786432 (1536x512)
    unsigned short* WihP = (unsigned short*)carve(wihN * 2);
    unsigned short* WhhP = (unsigned short*)carve(whhN * 2);
    unsigned short* Hq   = (unsigned short*)carve((size_t)E * 512 * 2);
    float*    a_buf   = (float*)carve((size_t)E * 8 * 4);
    float*    aexp    = (float*)carve((size_t)E * 8 * 4);
    unsigned* amaxEnc = (unsigned*)carve((size_t)ND * 8 * 4);
    float*    asum    = (float*)carve((size_t)ND * 8 * 4);
    (void)ws_size; (void)n_in;

    const int ndh = ND * 8;
    const int initN = out_size > ndh ? out_size : ndh;
    init_kernel<<<(initN + 255) / 256, 256, 0, stream>>>(out, amaxEnc, asum,
                                                         out_size, ndh);
    repack_kernel<<<((int)wihN + 255) / 256, 256, 0, stream>>>(W_ih, WihP, 64, 2,
                                                               (int)wihN);
    repack_kernel<<<((int)whhN + 255) / 256, 256, 0, stream>>>(W_hh, WhhP, 512, 16,
                                                               (int)whhN);
    gru_attn_kernel<<<(E + 31) / 32, 256, 0, stream>>>(features, WihP, WhhP, b_ih,
                                                       b_hh, attn, emi, dstv, Hq,
                                                       a_buf, amaxEnc, E, L);
    exp_sum_kernel<<<(E * 8 + 255) / 256, 256, 0, stream>>>(a_buf, amaxEnc, dstv,
                                                            aexp, asum, E * 8);
    unsigned aggBlocks = (unsigned)(((size_t)E * 512 + 255) / 256);
    aggregate_kernel<<<aggBlocks, 256, 0, stream>>>(Hq, aexp, asum, dstv, out, E);
}
